// HyperConnections_12463995093502
// MI455X (gfx1250) — compile-verified, matrix operand order checked by reading, not run on hardware
//
#include <hip/hip_runtime.h>

typedef __attribute__((ext_vector_type(16))) __bf16 v16bf;
typedef __attribute__((ext_vector_type(8)))  float  v8f;
typedef __attribute__((ext_vector_type(2)))  float  v2f;

#define BUS_ 8192
#define MM_ 64
#define CC_ 128
#define NBIG_ 4096
#define TOKENS_ 8192
#define TOK_PER_BLK 16

union ABFrag { uint4 q[2]; v16bf v; };
union US16   { unsigned short s[16]; uint4 q[2]; };

__device__ __forceinline__ unsigned short f2bf(float f) {
    unsigned int u = __float_as_uint(f);
    unsigned int r = u + 0x7FFFu + ((u >> 16) & 1u);   // round-to-nearest-even
    return (unsigned short)(r >> 16);
}

// K index inside a bf16 16x32 WMMA fragment: element v of lane (lo/hi half)
__device__ __forceinline__ int kidx(int laneHi, int v) {
    return laneHi ? (v < 8 ? v + 8 : v + 16) : (v < 8 ? v : v + 8);
}

// ---------------------------------------------------------------------------
// Kernel 1: per-token rmsnorm -> swizzled bf16 A fragments; H_pre; branch_input
// One 256-thread block per token. Thread t owns k-range [t*32, t*32+32) = kstep t.
// ---------------------------------------------------------------------------
__global__ __launch_bounds__(256) void hk_norm(
    const float* __restrict__ resid, const float* __restrict__ gamma,
    const float* __restrict__ phi_pre, const float* __restrict__ Hpre_logits,
    const float* __restrict__ alpha_pre,
    unsigned short* __restrict__ Asw, float* __restrict__ branch_out)
{
    const int token = blockIdx.x;
    const int tid = threadIdx.x;
    const float* x = resid + (size_t)token * BUS_;

    __shared__ float sred[256];
    __shared__ float pres[4];
    __shared__ float hpre[4];

    // pass 1: sum of squares
    float xl[32];
    float ss = 0.f;
    const int base = tid * 32;
#pragma unroll
    for (int v = 0; v < 32; ++v) { float t = x[base + v]; xl[v] = t; ss += t * t; }
    sred[tid] = ss; __syncthreads();
    for (int o = 128; o > 0; o >>= 1) { if (tid < o) sred[tid] += sred[tid + o]; __syncthreads(); }
    const float s = 90.50966799187809f / fmaxf(sqrtf(sred[0]), 1e-12f); // sqrt(8192)/||x||

    // pass 2: normed values, pre-projection partials
    float y[32];
    float pr[4] = {0.f, 0.f, 0.f, 0.f};
#pragma unroll
    for (int v = 0; v < 32; ++v) {
        const int i = base + v;
        const float yy = xl[v] * s * (gamma[i] + 1.0f);
        y[v] = yy;
        const float4 pp = *(const float4*)(phi_pre + (size_t)i * 4);
        pr[0] += yy * pp.x; pr[1] += yy * pp.y; pr[2] += yy * pp.z; pr[3] += yy * pp.w;
    }

    // swizzled bf16 A-fragment write: kstep = tid, token row tl within block tb
    {
        const int tb = token >> 4, tl = token & 15;
        US16 lo, hi;
#pragma unroll
        for (int v = 0; v < 16; ++v) {
            lo.s[v] = f2bf(y[kidx(0, v)]);
            hi.s[v] = f2bf(y[kidx(1, v)]);
        }
        const size_t abase = ((size_t)tb * 256 + tid) * 32;
        uint4* plo = (uint4*)(Asw + (abase + tl) * 16);
        uint4* phi = (uint4*)(Asw + (abase + tl + 16) * 16);
        plo[0] = lo.q[0]; plo[1] = lo.q[1];
        phi[0] = hi.q[0]; phi[1] = hi.q[1];
    }

    // reduce the 4 pre dots
    for (int kk = 0; kk < 4; ++kk) {
        __syncthreads();
        sred[tid] = pr[kk]; __syncthreads();
        for (int o = 128; o > 0; o >>= 1) { if (tid < o) sred[tid] += sred[tid + o]; __syncthreads(); }
        if (tid == 0) pres[kk] = sred[0];
    }
    __syncthreads();

    if (tid == 0) {
        const float a = alpha_pre[0];
        float l0 = a * pres[0] + Hpre_logits[0];
        float l1 = a * pres[1] + Hpre_logits[1];
        float l2 = a * pres[2] + Hpre_logits[2];
        float l3 = a * pres[3] + Hpre_logits[3];
        const float mx = fmaxf(fmaxf(l0, l1), fmaxf(l2, l3));
        const float e0 = __expf(l0 - mx), e1 = __expf(l1 - mx);
        const float e2 = __expf(l2 - mx), e3 = __expf(l3 - mx);
        const float inv = 1.0f / (e0 + e1 + e2 + e3);
        hpre[0] = e0 * inv; hpre[1] = e1 * inv; hpre[2] = e2 * inv; hpre[3] = e3 * inv;
    }
    __syncthreads();

    const float h0 = hpre[0], h1 = hpre[1], h2 = hpre[2], h3 = hpre[3];
    for (int d = tid; d < 2048; d += 256) {
        const float b = h0 * x[d] + h1 * x[2048 + d] + h2 * x[4096 + d] + h3 * x[6144 + d];
        branch_out[(size_t)token * 2048 + d] = b;
    }
}

// ---------------------------------------------------------------------------
// Kernel 2: phi_res fp32 [8192][4096] -> bf16 WMMA B-fragment layout
// One thread per (ntile, kstep, lane): writes 16 contiguous bf16 (32 B).
// ---------------------------------------------------------------------------
__global__ __launch_bounds__(256) void hk_swzB(
    const float* __restrict__ phi_res, unsigned short* __restrict__ Bsw)
{
    const int gid = blockIdx.x * 256 + threadIdx.x;   // ((nt*256)+ks)*32+lane
    const int lane = gid & 31;
    const int ks = (gid >> 5) & 255;
    const int nt = gid >> 13;
    if (nt >= 256) return;
    const int n = nt * 16 + (lane & 15);
    const int hi = lane >> 4;
    US16 out;
#pragma unroll
    for (int v = 0; v < 16; ++v) {
        const int k = ks * 32 + kidx(hi, v);
        out.s[v] = f2bf(phi_res[(size_t)k * NBIG_ + n]);
    }
    uint4* p = (uint4*)(Bsw + (size_t)gid * 16);
    p[0] = out.q[0]; p[1] = out.q[1];
}

// ---------------------------------------------------------------------------
// Kernel 3: fused bf16-WMMA GEMM -> logits in LDS -> Sinkhorn -> f32-WMMA mix
// 512 blocks x 256 threads (8 waves); 16 tokens per block; 270336 B dynamic LDS
// ---------------------------------------------------------------------------
__global__ __launch_bounds__(256) void hk_main(
    const unsigned short* __restrict__ Asw, const unsigned short* __restrict__ Bsw,
    const float* __restrict__ resid, const float* __restrict__ gamma,
    const float* __restrict__ phi_post, const float* __restrict__ Hres,
    const float* __restrict__ Hpost_logits, const float* __restrict__ alpha_res,
    const float* __restrict__ alpha_post,
    float* __restrict__ mixed_out, float* __restrict__ hpost_out)
{
    extern __shared__ float lds[];                 // [16*4096] logits/P + [16*128] u,v
    float* uvb = lds + TOK_PER_BLK * NBIG_;

    const int tb = blockIdx.x;                     // token block (16 tokens)
    const int tid = threadIdx.x;
    const int wave = tid >> 5, lane = tid & 31;
    const int laneHi = lane >> 4, ln16 = lane & 15;
    const float ares = alpha_res[0];

    // ---------------- Phase 1: 16 x 4096 x 8192 bf16 WMMA GEMM ----------------
    for (int pass = 0; pass < 2; ++pass) {
        v8f acc[16];
#pragma unroll
        for (int i = 0; i < 16; ++i) acc[i] = (v8f)0.0f;

        const unsigned short* aptr = Asw + ((size_t)tb * 256 * 32 + lane) * 16;
        for (int ks = 0; ks < 256; ++ks) {
            ABFrag af;
            af.q[0] = *(const uint4*)(aptr);
            af.q[1] = *(const uint4*)(aptr + 8);
            aptr += 32 * 16;
#pragma unroll
            for (int i = 0; i < 16; ++i) {
                const int nt = wave + 8 * (pass * 16 + i);
                const unsigned short* bptr = Bsw + (((size_t)nt * 256 + ks) * 32 + lane) * 16;
                ABFrag bfr;
                bfr.q[0] = *(const uint4*)(bptr);
                bfr.q[1] = *(const uint4*)(bptr + 8);
                acc[i] = __builtin_amdgcn_wmma_f32_16x16x32_bf16(
                    false, af.v, false, bfr.v, (short)0, acc[i], false, false);
            }
        }
        // scaled logits -> LDS (logit = alpha_res*raw + H_res[i][j])
#pragma unroll
        for (int i = 0; i < 16; ++i) {
            const int nt = wave + 8 * (pass * 16 + i);
            const int nbig = nt * 16 + ln16;
            const float h = Hres[nbig];
#pragma unroll
            for (int r = 0; r < 8; ++r) {
                const int trow = r + (laneHi ? 8 : 0);
                lds[trow * NBIG_ + nbig] = ares * acc[i][r] + h;
            }
        }
    }
    __syncthreads();

    // ---------------- Phase 2+3: per-token Sinkhorn + WMMA f32 mix ----------------
    const float LOGR = -4.1588830833596715f;       // -log(64)
    const float apost = alpha_post[0];

    for (int tk = 0; tk < 2; ++tk) {
        const int t_local = wave * 2 + tk;
        float* Z = lds + t_local * NBIG_;
        float* u = uvb + t_local * 128;
        float* v = u + 64;
        v[lane] = 0.f; v[lane + 32] = 0.f;
        __syncthreads();

        for (int it = 0; it < 20; ++it) {
#pragma unroll
            for (int rr = 0; rr < 2; ++rr) {       // u update: rows lane, lane+32
                const int i = lane + rr * 32;
                float m = -1e30f;
#pragma unroll 8
                for (int j = 0; j < 64; ++j) m = fmaxf(m, Z[i * 64 + j] + v[j]);
                float sum = 0.f;
#pragma unroll 8
                for (int j = 0; j < 64; ++j) sum += __expf(Z[i * 64 + j] + v[j] - m);
                u[i] = LOGR - (m + __logf(sum));
            }
            __syncthreads();
#pragma unroll
            for (int cc = 0; cc < 2; ++cc) {       // v update: cols lane, lane+32
                const int j = lane + cc * 32;
                float m = -1e30f;
#pragma unroll 8
                for (int i2 = 0; i2 < 64; ++i2) m = fmaxf(m, Z[i2 * 64 + j] + u[i2]);
                float sum = 0.f;
#pragma unroll 8
                for (int i2 = 0; i2 < 64; ++i2) sum += __expf(Z[i2 * 64 + j] + u[i2] - m);
                v[j] = LOGR - (m + __logf(sum));
            }
            __syncthreads();
        }

        // P = exp(Z + u + v) * rows, in place
#pragma unroll
        for (int rr = 0; rr < 2; ++rr) {
            const int i = lane + rr * 32;
            const float ui = u[i];
#pragma unroll 8
            for (int j = 0; j < 64; ++j)
                Z[i * 64 + j] = __expf(Z[i * 64 + j] + ui + v[j]) * 64.0f;
        }
        __syncthreads();

        // mixed = P(64x64) @ reshaped(64x128) via v_wmma_f32_16x16x4_f32
        const int token = tb * TOK_PER_BLK + t_local;
        const float* rsrc = resid + (size_t)token * BUS_;
        float* mout = mixed_out + (size_t)token * BUS_;
        float ss = 0.f, d0 = 0.f, d1 = 0.f, d2 = 0.f, d3 = 0.f;

        for (int ti = 0; ti < 4; ++ti) {
            for (int tc = 0; tc < 8; ++tc) {
                v8f mac = (v8f)0.0f;
#pragma unroll
                for (int kq = 0; kq < 16; ++kq) {
                    const int k0 = kq * 4 + (laneHi ? 2 : 0);
                    v2f av, bv;
                    av[0] = Z[(ti * 16 + ln16) * 64 + k0];
                    av[1] = Z[(ti * 16 + ln16) * 64 + k0 + 1];
                    bv[0] = rsrc[(size_t)k0 * CC_ + tc * 16 + ln16];
                    bv[1] = rsrc[(size_t)(k0 + 1) * CC_ + tc * 16 + ln16];
                    mac = __builtin_amdgcn_wmma_f32_16x16x4_f32(
                        false, av, false, bv, (short)0, mac, false, false);
                }
#pragma unroll
                for (int r = 0; r < 8; ++r) {
                    const int row = ti * 16 + r + (laneHi ? 8 : 0);
                    const int col = tc * 16 + ln16;
                    const int bi = row * CC_ + col;
                    const float val = mac[r];
                    mout[bi] = val;
                    ss += val * val;
                    const float vg = val * (gamma[bi] + 1.0f);
                    const float4 pp = *(const float4*)(phi_post + (size_t)bi * 4);
                    d0 += vg * pp.x; d1 += vg * pp.y; d2 += vg * pp.z; d3 += vg * pp.w;
                }
            }
        }

        // wave reduce -> H_post softmax
        for (int o = 16; o >= 1; o >>= 1) {
            ss += __shfl_xor(ss, o, 32);
            d0 += __shfl_xor(d0, o, 32);
            d1 += __shfl_xor(d1, o, 32);
            d2 += __shfl_xor(d2, o, 32);
            d3 += __shfl_xor(d3, o, 32);
        }
        if (lane == 0) {
            const float sc = 90.50966799187809f / fmaxf(sqrtf(ss), 1e-12f);
            float l0 = apost * sc * d0 + Hpost_logits[0];
            float l1 = apost * sc * d1 + Hpost_logits[1];
            float l2 = apost * sc * d2 + Hpost_logits[2];
            float l3 = apost * sc * d3 + Hpost_logits[3];
            const float mx = fmaxf(fmaxf(l0, l1), fmaxf(l2, l3));
            const float e0 = __expf(l0 - mx), e1 = __expf(l1 - mx);
            const float e2 = __expf(l2 - mx), e3 = __expf(l3 - mx);
            const float inv = 1.0f / (e0 + e1 + e2 + e3);
            hpost_out[(size_t)token * 4 + 0] = e0 * inv;
            hpost_out[(size_t)token * 4 + 1] = e1 * inv;
            hpost_out[(size_t)token * 4 + 2] = e2 * inv;
            hpost_out[(size_t)token * 4 + 3] = e3 * inv;
        }
        __syncthreads();
    }
}

// ---------------------------------------------------------------------------
extern "C" void kernel_launch(void* const* d_in, const int* in_sizes, int n_in,
                              void* d_out, int out_size, void* d_ws, size_t ws_size,
                              hipStream_t stream) {
    const float* residuals = (const float*)d_in[0];
    const float* gamma     = (const float*)d_in[1];
    const float* phi_res   = (const float*)d_in[2];
    const float* phi_pre   = (const float*)d_in[3];
    const float* phi_post  = (const float*)d_in[4];
    const float* Hres      = (const float*)d_in[5];
    const float* Hpre      = (const float*)d_in[6];
    const float* Hpost     = (const float*)d_in[7];
    const float* a_res     = (const float*)d_in[8];
    const float* a_pre     = (const float*)d_in[9];
    const float* a_post    = (const float*)d_in[10];

    float* branch = (float*)d_out;                               // [2*4096*2048]
    float* mixed  = branch + (size_t)2 * 4096 * 2048;            // [2*4096*8192]
    float* hpost  = mixed  + (size_t)2 * 4096 * 8192;            // [2*4096*4]

    unsigned short* Asw = (unsigned short*)d_ws;                 // 128 MB
    unsigned short* Bsw = Asw + (size_t)TOKENS_ * BUS_;          // 64 MB

    hk_norm<<<TOKENS_, 256, 0, stream>>>(residuals, gamma, phi_pre, Hpre, a_pre,
                                         Asw, branch);
    hk_swzB<<<8192, 256, 0, stream>>>(phi_res, Bsw);

    const size_t ldsBytes = (TOK_PER_BLK * NBIG_ + TOK_PER_BLK * 128) * sizeof(float); // 270336
    hk_main<<<TOKENS_ / TOK_PER_BLK, 256, ldsBytes, stream>>>(
        Asw, Bsw, residuals, gamma, phi_post, Hres, Hpost, a_res, a_post,
        mixed, hpost);
}
